// PCSR_61100204753040
// MI455X (gfx1250) — compile-verified, matrix operand order checked by reading, not run on hardware
//
#include <hip/hip_runtime.h>
#include <hip/hip_bf16.h>

// ---------------------------------------------------------------------------
// Problem constants (from the reference)
// ---------------------------------------------------------------------------
#define HL 96
#define WL 96
#define HH 384
#define WH 384
#define QN (HH * WH)        // 147456
#define CF 64               // encoder channels
#define KIN 96              // IN_DIM=68 padded to multiple of 32
#define NPAD 16             // small output dims padded to 16
#define CHUNK (QN / 4)      // 36864 rows per heavy-MLP chunk

typedef __attribute__((ext_vector_type(16))) _Float16 v16h;
typedef __attribute__((ext_vector_type(8)))  float    v8f;

// ---------------------------------------------------------------------------
// Generic WMMA GEMM:  C[M,N] = act(A[M,K] * BT[N,K]^T + bias[N])
// A, BT are f16 row-major with K a multiple of 32; M mult of 16, N mult of
// 16*NT.  One wave computes a 16 x (16*NT) strip.  Per K-step: ALL fragment
// loads (A + NT x B, from NT independent precomputed row pointers) are issued
// before any WMMA so the scheduler can clause them and stage s_wait_loadcnt,
// overlapping VMEM latency with the 4 independent v_wmma_f32_16x16x32_f16
// accumulator chains.  Fragment layouts follow CDNA5 ISA 7.12.2.
// ---------------------------------------------------------------------------
template <bool RELU, int NT, typename OutT>
__global__ void wmma_gemm_kernel(const _Float16* __restrict__ A,
                                 const _Float16* __restrict__ BT,
                                 const float* __restrict__ bias,
                                 OutT* __restrict__ C,
                                 int M, int N, int K) {
    const int wave  = blockIdx.x * (blockDim.x >> 5) + (threadIdx.x >> 5);
    const int ngrp  = N / (16 * NT);
    const int tiles = (M >> 4) * ngrp;
    if (wave >= tiles) return;                 // wave-granular: EXEC stays all-1s

    const int mt   = wave / ngrp;
    const int ng   = wave % ngrp;
    const int lane = threadIdx.x & 31;
    const int lrow = lane & 15;                // M-row for A frag / N-row for B frag
    const int half = lane >> 4;                // K-half selector

    const _Float16* arow = A + (size_t)(mt * 16 + lrow) * K + half * 8;
    const _Float16* brow[NT];
#pragma unroll
    for (int t = 0; t < NT; ++t)
        brow[t] = BT + (size_t)((ng * NT + t) * 16 + lrow) * K + half * 8;

    v8f acc[NT];
#pragma unroll
    for (int t = 0; t < NT; ++t) acc[t] = (v8f){};

    for (int k0 = 0; k0 < K; k0 += 32) {
        if (k0 + 32 < K) __builtin_prefetch(arow + k0 + 32, 0, 1);  // global_prefetch_b8

        union { v16h v; unsigned int u[8]; } a, b[NT];
        // Issue ALL loads of this K-step first (clause + staged waits).
#pragma unroll
        for (int r = 0; r < 8; ++r) {
            // k offset within the 16x32 tile: (r/4)*16 + (r&3)*2 (+ half*8 baked in)
            const int k = k0 + ((r >> 2) << 4) + ((r & 3) << 1);
            a.u[r] = *(const unsigned int*)(arow + k);
        }
#pragma unroll
        for (int t = 0; t < NT; ++t) {
#pragma unroll
            for (int r = 0; r < 8; ++r) {
                const int k = k0 + ((r >> 2) << 4) + ((r & 3) << 1);
                b[t].u[r] = *(const unsigned int*)(brow[t] + k);
            }
        }
        // Then the NT independent WMMA accumulation chains.
#pragma unroll
        for (int t = 0; t < NT; ++t) {
            acc[t] = __builtin_amdgcn_wmma_f32_16x16x32_f16(
                /*neg_a=*/false, a.v, /*neg_b=*/false, b[t].v,
                /*c_mod=*/(short)0, acc[t], /*reuse_a=*/false, /*reuse_b=*/false);
        }
    }

#pragma unroll
    for (int t = 0; t < NT; ++t) {
        const int   ncol = (ng * NT + t) * 16 + lrow;
        const float bv   = bias[ncol];
#pragma unroll
        for (int r = 0; r < 8; ++r) {
            const int mrow = mt * 16 + half * 8 + r;  // C/D: VGPR r -> M=r (+8 hi lanes)
            float v = acc[t][r] + bv;
            if (RELU) v = v > 0.f ? v : 0.f;
            C[(size_t)mrow * N + ncol] = (OutT)v;
        }
    }
}

// ---------------------------------------------------------------------------
// 3x3 SAME conv (zero pad), NCHW, 96x96, CO=64, optional ReLU
// ---------------------------------------------------------------------------
__global__ void conv3x3_kernel(const float* __restrict__ in,
                               const float* __restrict__ w,
                               const float* __restrict__ b,
                               float* __restrict__ out,
                               int CI, int relu) {
    const int tid = blockIdx.x * blockDim.x + threadIdx.x;
    if (tid >= CF * HL * WL) return;
    const int co  = tid / (HL * WL);
    const int pos = tid - co * (HL * WL);
    const int y   = pos / WL;
    const int x   = pos - y * WL;

    float acc = b[co];
    for (int ci = 0; ci < CI; ++ci) {
        const float* ip = in + ci * (HL * WL);
        const float* wp = w + ((co * CI + ci) * 9);
#pragma unroll
        for (int ky = 0; ky < 3; ++ky) {
            const int iy = y + ky - 1;
            if (iy < 0 || iy >= HL) continue;
#pragma unroll
            for (int kx = 0; kx < 3; ++kx) {
                const int ix = x + kx - 1;
                if (ix < 0 || ix >= WL) continue;
                acc += ip[iy * WL + ix] * wp[ky * 3 + kx];
            }
        }
    }
    if (relu && acc < 0.f) acc = 0.f;
    out[tid] = acc;
}

// ---------------------------------------------------------------------------
// Pack weight [IN,OUT] f32 -> WT [Np,Kp] f16 (transpose + zero pad)
// ---------------------------------------------------------------------------
__global__ void pack_wt_kernel(const float* __restrict__ w, _Float16* __restrict__ wt,
                               int IN, int OUT, int Kp, int Np) {
    const int t = blockIdx.x * blockDim.x + threadIdx.x;
    if (t >= Np * Kp) return;
    const int n = t / Kp;
    const int k = t - n * Kp;
    wt[t] = (n < OUT && k < IN) ? (_Float16)w[(size_t)k * OUT + n] : (_Float16)0.f;
}

__global__ void pack_bias_kernel(const float* __restrict__ b, float* __restrict__ bp,
                                 int OUT, int Np) {
    const int t = blockIdx.x * blockDim.x + threadIdx.x;
    if (t >= Np) return;
    bp[t] = (t < OUT) ? b[t] : 0.f;
}

// ---------------------------------------------------------------------------
// Build per-query MLP input: nearest feature gather + rel coord + scaled cell
// inp[q, 0..63]=feat, [64..65]=rel*(h,w), [66..67]=cell*(h,w), [68..95]=0
// ---------------------------------------------------------------------------
__global__ void build_inp_kernel(const float* __restrict__ feat,
                                 const float* __restrict__ coord,
                                 const float* __restrict__ cell,
                                 _Float16* __restrict__ inp) {
    const int q = blockIdx.x * blockDim.x + threadIdx.x;
    if (q >= QN) return;
    const float cy = coord[2 * q + 0];
    const float cx = coord[2 * q + 1];
    float fyi = floorf((cy + 1.f) * 0.5f * (float)HL);
    float fxi = floorf((cx + 1.f) * 0.5f * (float)WL);
    fyi = fminf(fmaxf(fyi, 0.f), (float)(HL - 1));
    fxi = fminf(fmaxf(fxi, 0.f), (float)(WL - 1));
    const int iy = (int)fyi;
    const int ix = (int)fxi;
    const int idx = iy * WL + ix;

    _Float16* o = inp + (size_t)q * KIN;
#pragma unroll
    for (int c = 0; c < CF; ++c) o[c] = (_Float16)feat[c * (HL * WL) + idx];

    const float fy = -1.f + (2.f * (float)iy + 1.f) / (float)HL;
    const float fx = -1.f + (2.f * (float)ix + 1.f) / (float)WL;
    o[64] = (_Float16)((cy - fy) * (float)HL);
    o[65] = (_Float16)((cx - fx) * (float)WL);
    o[66] = (_Float16)(cell[2 * q + 0] * (float)HL);
    o[67] = (_Float16)(cell[2 * q + 1] * (float)WL);
#pragma unroll
    for (int k = 68; k < KIN; ++k) o[k] = (_Float16)0.f;
}

// ---------------------------------------------------------------------------
// Fuse: flag = argmax(classifier logits); pred = (flag? heavy : light) +
// bilinear-border sample of LR image.  Writes p as [3, H*W] + flag buffer.
// ---------------------------------------------------------------------------
__global__ void fuse_kernel(const float* __restrict__ lr,
                            const float* __restrict__ coord,
                            const float* __restrict__ cls,
                            const float* __restrict__ lgt,
                            const float* __restrict__ hvy,
                            float* __restrict__ pbuf,
                            int* __restrict__ flagbuf) {
    const int q = blockIdx.x * blockDim.x + threadIdx.x;
    if (q >= QN) return;
    const int flag = cls[(size_t)q * NPAD + 1] > cls[(size_t)q * NPAD + 0] ? 1 : 0;

    const float cy = coord[2 * q + 0];
    const float cx = coord[2 * q + 1];
    const float fy = fminf(fmaxf((cy + 1.f) * (0.5f * (float)HL) - 0.5f, 0.f), (float)(HL - 1));
    const float fx = fminf(fmaxf((cx + 1.f) * (0.5f * (float)WL) - 0.5f, 0.f), (float)(WL - 1));
    const int y0 = (int)floorf(fy);
    const int x0 = (int)floorf(fx);
    const float wy = fy - (float)y0;
    const float wx = fx - (float)x0;
    const int y1 = (y0 + 1 < HL) ? y0 + 1 : HL - 1;
    const int x1 = (x0 + 1 < WL) ? x0 + 1 : WL - 1;

#pragma unroll
    for (int c = 0; c < 3; ++c) {
        const float* img = lr + c * (HL * WL);
        const float bil = img[y0 * WL + x0] * (1.f - wy) * (1.f - wx)
                        + img[y0 * WL + x1] * (1.f - wy) * wx
                        + img[y1 * WL + x0] * wy * (1.f - wx)
                        + img[y1 * WL + x1] * wy * wx;
        const float pr = flag ? hvy[(size_t)q * NPAD + c] : lgt[(size_t)q * NPAD + c];
        pbuf[(size_t)c * QN + q] = pr + bil;
    }
    flagbuf[q] = flag;
}

// ---------------------------------------------------------------------------
// Refinement: interior easy pixels adjacent to hard pixels -> 3x3 mean.
// Writes final output: pred [Q,3] then flag [Q] (as float).
// ---------------------------------------------------------------------------
__global__ void refine_kernel(const float* __restrict__ pbuf,
                              const int* __restrict__ flagbuf,
                              float* __restrict__ out) {
    const int q = blockIdx.x * blockDim.x + threadIdx.x;
    if (q >= QN) return;
    const int y = q / WH;
    const int x = q - y * WH;
    const int fl = flagbuf[q];

    float v0 = pbuf[q];
    float v1 = pbuf[QN + q];
    float v2 = pbuf[2 * (size_t)QN + q];

    const bool interior = (y > 0) && (y < HH - 1) && (x > 0) && (x < WH - 1);
    if (fl == 0 && interior) {
        int fsum = 0;
        float s0 = 0.f, s1 = 0.f, s2 = 0.f;
#pragma unroll
        for (int dy = -1; dy <= 1; ++dy) {
#pragma unroll
            for (int dx = -1; dx <= 1; ++dx) {
                const int qq = (y + dy) * WH + (x + dx);
                fsum += flagbuf[qq];
                s0 += pbuf[qq];
                s1 += pbuf[QN + qq];
                s2 += pbuf[2 * (size_t)QN + qq];
            }
        }
        if (fsum > 0) {
            v0 = s0 * (1.f / 9.f);
            v1 = s1 * (1.f / 9.f);
            v2 = s2 * (1.f / 9.f);
        }
    }
    out[(size_t)q * 3 + 0] = v0;
    out[(size_t)q * 3 + 1] = v1;
    out[(size_t)q * 3 + 2] = v2;
    out[3 * (size_t)QN + q] = (float)fl;
}

// ---------------------------------------------------------------------------
// Host launch
// ---------------------------------------------------------------------------
extern "C" void kernel_launch(void* const* d_in, const int* in_sizes, int n_in,
                              void* d_out, int out_size, void* d_ws, size_t ws_size,
                              hipStream_t stream) {
    (void)in_sizes; (void)n_in; (void)out_size; (void)ws_size;

    // Inputs in setup_inputs() order
    const float* lr    = (const float*)d_in[0];
    const float* coord = (const float*)d_in[1];
    const float* cell  = (const float*)d_in[2];
    const float* ew1 = (const float*)d_in[3];  const float* eb1 = (const float*)d_in[4];
    const float* ew2 = (const float*)d_in[5];  const float* eb2 = (const float*)d_in[6];
    const float* ew3 = (const float*)d_in[7];  const float* eb3 = (const float*)d_in[8];
    const float* wh1 = (const float*)d_in[9];  const float* bh1 = (const float*)d_in[10];
    const float* wh2 = (const float*)d_in[11]; const float* bh2 = (const float*)d_in[12];
    const float* wh3 = (const float*)d_in[13]; const float* bh3 = (const float*)d_in[14];
    const float* wh4 = (const float*)d_in[15]; const float* bh4 = (const float*)d_in[16];
    const float* wl1 = (const float*)d_in[17]; const float* bl1 = (const float*)d_in[18];
    const float* wl2 = (const float*)d_in[19]; const float* bl2 = (const float*)d_in[20];
    const float* wc1 = (const float*)d_in[21]; const float* bc1 = (const float*)d_in[22];
    const float* wc2 = (const float*)d_in[23]; const float* bc2 = (const float*)d_in[24];

    float* out = (float*)d_out;

    // Workspace carving (256B aligned)
    size_t off = 0;
    char* base = (char*)d_ws;
    auto alloc = [&](size_t bytes) -> void* {
        void* p = base + off;
        off += (bytes + 255) & ~(size_t)255;
        return p;
    };

    float*     featA   = (float*)alloc((size_t)CF * HL * WL * 4);
    float*     featB   = (float*)alloc((size_t)CF * HL * WL * 4);
    float*     featC   = (float*)alloc((size_t)CF * HL * WL * 4);
    _Float16*  inp16   = (_Float16*)alloc((size_t)QN * KIN * 2);
    _Float16*  bufA    = (_Float16*)alloc((size_t)CHUNK * 256 * 2);  // also reused as [QN,64]
    _Float16*  bufB    = (_Float16*)alloc((size_t)CHUNK * 256 * 2);
    float*     clsOut  = (float*)alloc((size_t)QN * NPAD * 4);
    float*     lgtOut  = (float*)alloc((size_t)QN * NPAD * 4);
    float*     hvyOut  = (float*)alloc((size_t)QN * NPAD * 4);
    float*     pbuf    = (float*)alloc((size_t)3 * QN * 4);
    int*       flagbuf = (int*)alloc((size_t)QN * 4);
    _Float16*  wh1T = (_Float16*)alloc(256 * KIN * 2);
    _Float16*  wh2T = (_Float16*)alloc(256 * 256 * 2);
    _Float16*  wh3T = (_Float16*)alloc(256 * 256 * 2);
    _Float16*  wh4T = (_Float16*)alloc(NPAD * 256 * 2);
    _Float16*  wl1T = (_Float16*)alloc(64 * KIN * 2);
    _Float16*  wl2T = (_Float16*)alloc(NPAD * 64 * 2);
    _Float16*  wc1T = (_Float16*)alloc(64 * KIN * 2);
    _Float16*  wc2T = (_Float16*)alloc(NPAD * 64 * 2);
    float* bh1p = (float*)alloc(256 * 4);
    float* bh2p = (float*)alloc(256 * 4);
    float* bh3p = (float*)alloc(256 * 4);
    float* bh4p = (float*)alloc(NPAD * 4);
    float* bl1p = (float*)alloc(64 * 4);
    float* bl2p = (float*)alloc(NPAD * 4);
    float* bc1p = (float*)alloc(64 * 4);
    float* bc2p = (float*)alloc(NPAD * 4);

    const int TB = 256;

    // --- Encoder convs ---
    const int convN = CF * HL * WL;
    conv3x3_kernel<<<(convN + TB - 1) / TB, TB, 0, stream>>>(lr,    ew1, eb1, featA, 3,  1);
    conv3x3_kernel<<<(convN + TB - 1) / TB, TB, 0, stream>>>(featA, ew2, eb2, featB, CF, 1);
    conv3x3_kernel<<<(convN + TB - 1) / TB, TB, 0, stream>>>(featB, ew3, eb3, featC, CF, 0);

    // --- Weight packing ---
    auto packw = [&](const float* w, _Float16* wt, int IN, int OUT, int Kp, int Np) {
        const int n = Np * Kp;
        pack_wt_kernel<<<(n + TB - 1) / TB, TB, 0, stream>>>(w, wt, IN, OUT, Kp, Np);
    };
    auto packb = [&](const float* b, float* bp, int OUT, int Np) {
        pack_bias_kernel<<<(Np + TB - 1) / TB, TB, 0, stream>>>(b, bp, OUT, Np);
    };
    packw(wh1, wh1T, 68, 256, KIN, 256);  packb(bh1, bh1p, 256, 256);
    packw(wh2, wh2T, 256, 256, 256, 256); packb(bh2, bh2p, 256, 256);
    packw(wh3, wh3T, 256, 256, 256, 256); packb(bh3, bh3p, 256, 256);
    packw(wh4, wh4T, 256, 3, 256, NPAD);  packb(bh4, bh4p, 3, NPAD);
    packw(wl1, wl1T, 68, 64, KIN, 64);    packb(bl1, bl1p, 64, 64);
    packw(wl2, wl2T, 64, 3, 64, NPAD);    packb(bl2, bl2p, 3, NPAD);
    packw(wc1, wc1T, 68, 64, KIN, 64);    packb(bc1, bc1p, 64, 64);
    packw(wc2, wc2T, 64, 2, 64, NPAD);    packb(bc2, bc2p, 2, NPAD);

    // --- Per-query input build ---
    build_inp_kernel<<<(QN + TB - 1) / TB, TB, 0, stream>>>(featC, coord, cell, inp16);

    // --- WMMA GEMMs ---
    auto gemm_blocks = [&](int M, int N, int NT) {
        const int tiles = (M >> 4) * (N / (16 * NT));
        return (tiles + 7) / 8;   // 8 waves / 256-thread block
    };

    // Classifier: 96 -> 64 (relu) -> 16(2) logits
    wmma_gemm_kernel<true, 4, _Float16><<<gemm_blocks(QN, 64, 4), TB, 0, stream>>>(
        inp16, wc1T, bc1p, bufA, QN, 64, KIN);
    wmma_gemm_kernel<false, 1, float><<<gemm_blocks(QN, NPAD, 1), TB, 0, stream>>>(
        bufA, wc2T, bc2p, clsOut, QN, NPAD, 64);

    // Light head: 96 -> 64 (relu) -> 16(3)
    wmma_gemm_kernel<true, 4, _Float16><<<gemm_blocks(QN, 64, 4), TB, 0, stream>>>(
        inp16, wl1T, bl1p, bufA, QN, 64, KIN);
    wmma_gemm_kernel<false, 1, float><<<gemm_blocks(QN, NPAD, 1), TB, 0, stream>>>(
        bufA, wl2T, bl2p, lgtOut, QN, NPAD, 64);

    // Heavy head, 4 row-chunks ping-ponging bufA/bufB
    for (int ch = 0; ch < 4; ++ch) {
        const _Float16* aIn = inp16 + (size_t)ch * CHUNK * KIN;
        float* hOut = hvyOut + (size_t)ch * CHUNK * NPAD;
        wmma_gemm_kernel<true, 4, _Float16><<<gemm_blocks(CHUNK, 256, 4), TB, 0, stream>>>(
            aIn, wh1T, bh1p, bufA, CHUNK, 256, KIN);
        wmma_gemm_kernel<true, 4, _Float16><<<gemm_blocks(CHUNK, 256, 4), TB, 0, stream>>>(
            bufA, wh2T, bh2p, bufB, CHUNK, 256, 256);
        wmma_gemm_kernel<true, 4, _Float16><<<gemm_blocks(CHUNK, 256, 4), TB, 0, stream>>>(
            bufB, wh3T, bh3p, bufA, CHUNK, 256, 256);
        wmma_gemm_kernel<false, 1, float><<<gemm_blocks(CHUNK, NPAD, 1), TB, 0, stream>>>(
            bufA, wh4T, bh4p, hOut, CHUNK, NPAD, 256);
    }

    // --- Select + bilinear base, then box refinement ---
    fuse_kernel<<<(QN + TB - 1) / TB, TB, 0, stream>>>(
        lr, coord, clsOut, lgtOut, hvyOut, pbuf, flagbuf);
    refine_kernel<<<(QN + TB - 1) / TB, TB, 0, stream>>>(pbuf, flagbuf, out);
}
